// Attention_20212116095153
// MI455X (gfx1250) — matrix operand order repaired and checked by reading, not compiled, not run on hardware
//
#include <hip/hip_runtime.h>
#include <hip/hip_bf16.h>
#include <math.h>

// ---------------------------------------------------------------------------
// Types for CDNA5 WMMA (wave32): v16bf A/B fragments, v8f C/D accumulators.
// ---------------------------------------------------------------------------
typedef __attribute__((ext_vector_type(16))) __bf16 v16bf;
typedef __attribute__((ext_vector_type(8)))  float  v8f;
typedef __attribute__((ext_vector_type(4)))  float  f4raw;   // 16-byte move unit
typedef __attribute__((ext_vector_type(4)))  unsigned int u32x4;
typedef __attribute__((ext_vector_type(4)))  int i32x4;
typedef __attribute__((ext_vector_type(8)))  int i32x8;

union FragU { f4raw f4[2]; v16bf v; };   // 32 bytes, trivially constructible

#define D_MODEL  1024
#define N_HEADS  16
#define HEAD_DIM 64
#define SEQ      2048
#define BATCH    4
#define MTOT     (BATCH * SEQ)   // 8192 rows

#if defined(__has_builtin)
#if __has_builtin(__builtin_amdgcn_tensor_load_to_lds) && __has_builtin(__builtin_amdgcn_s_wait_tensorcnt)
#define HAVE_TDM 1
#endif
#endif
#ifndef HAVE_TDM
#define HAVE_TDM 0
#endif

__device__ __forceinline__ unsigned short f32_to_bf16(float f) {
    union { float f; unsigned u; } x; x.f = f;
    unsigned u = x.u;
    unsigned r = u + 0x7FFFu + ((u >> 16) & 1u);           // round-to-nearest-even
    if ((u & 0x7F800000u) == 0x7F800000u && (u & 0x007FFFFFu)) r = u; // keep NaN
    return (unsigned short)(r >> 16);
}

#if HAVE_TDM
// LDS byte offset of a generic pointer to __shared__ (aperture: addr[31:0] = offset)
__device__ __forceinline__ unsigned lds_off(const void* p) {
    return (unsigned)(unsigned long long)p;
}

// TDM 2D tile load: tile_x contiguous elements (2B each) x tile_y rows,
// row stride = stride_x elements. Descriptor per CDNA5 ISA 8.3/8.4.
__device__ __forceinline__ void tdm_load_2d(unsigned ldsByteOff, unsigned long long gaddr,
                                            unsigned tile_x, unsigned tile_y,
                                            unsigned long long stride_x) {
    u32x4 g0;
    g0[0] = 1u;                                            // count=1, user descriptor
    g0[1] = ldsByteOff;                                    // lds_addr [63:32]
    g0[2] = (unsigned)(gaddr & 0xFFFFFFFFu);               // global_addr [95:64]
    g0[3] = (unsigned)((gaddr >> 32) & 0x01FFFFFFu)        // global_addr [120:96]
          | (2u << 30);                                    // type=2 ("image")
    i32x8 g1;
    g1[0] = (int)(1u << 16);                               // data_size=1 (2 bytes)
    g1[1] = (int)((tile_x & 0xFFFFu) << 16);               // tensor_dim0[15:0] (== tile_x)
    g1[2] = (int)((tile_x >> 16) | ((tile_y & 0xFFFFu) << 16)); // dim0 hi | tensor_dim1 lo
    g1[3] = (int)((tile_y >> 16) | ((tile_x & 0xFFFFu) << 16)); // dim1 hi | tile_dim0
    g1[4] = (int)(tile_y & 0xFFFFu);                       // tile_dim1 (tile_dim2=0)
    g1[5] = (int)(stride_x & 0xFFFFFFFFu);                 // tensor_dim0_stride lo32
    g1[6] = (int)((stride_x >> 32) & 0xFFFFu);             // stride hi16 (dim1_stride=0)
    g1[7] = 0;
    i32x4 gz = {0, 0, 0, 0};
#if __clang_major__ >= 23
    i32x8 gz8 = {0, 0, 0, 0, 0, 0, 0, 0};
    __builtin_amdgcn_tensor_load_to_lds(g0, g1, gz, gz, gz8, 0);
#else
    __builtin_amdgcn_tensor_load_to_lds(g0, g1, gz, gz, 0);
#endif
}
#endif // HAVE_TDM

// ---------------------------------------------------------------------------
// Stage 0: f32 -> bf16 cast (vectorized x4)
// ---------------------------------------------------------------------------
__global__ void cast_f32_bf16(const float* __restrict__ src,
                              unsigned short* __restrict__ dst, int n4) {
    int i = blockIdx.x * blockDim.x + threadIdx.x;
    if (i >= n4) return;
    f4raw v = *(const f4raw*)&src[i * 4];
    unsigned short o0 = f32_to_bf16(v.x), o1 = f32_to_bf16(v.y);
    unsigned short o2 = f32_to_bf16(v.z), o3 = f32_to_bf16(v.w);
    unsigned long long p = (unsigned long long)o0 | ((unsigned long long)o1 << 16) |
                           ((unsigned long long)o2 << 32) | ((unsigned long long)o3 << 48);
    *(unsigned long long*)&dst[i * 4] = p;
}

// ---------------------------------------------------------------------------
// GEMM: C[M,N] = A[M,K] * Bw[N,K]^T   (A,Bw bf16 row-major; acc f32)
// Block = 256 thr (8 waves), tile 128x128, K-step 32.
// Global->LDS staging via the Tensor Data Mover, double buffered:
// issue DMA for tile k+1, run WMMAs on tile k, s_wait_tensorcnt, barrier.
// ---------------------------------------------------------------------------
template <bool OUT_F32>
__global__ __launch_bounds__(256)
void gemm_bf16_wmma(const unsigned short* __restrict__ A,
                    const unsigned short* __restrict__ Bw,
                    void* __restrict__ C, int M, int N, int K) {
    __shared__ alignas(16) unsigned short sA[2][128 * 32];  // [row][k] k-contiguous
    __shared__ alignas(16) unsigned short sB[2][128 * 32];  // [n][k]   k-contiguous

    const int tid  = threadIdx.x;
    const int lane = tid & 31, wave = tid >> 5;
    const int half = lane >> 4, l16 = lane & 15;
    const int m0 = blockIdx.y * 128, n0 = blockIdx.x * 128;
    const int wm = wave & 1, wn = wave >> 1;
    const int NT = K / 32;

    v8f acc[4][2];
    #pragma unroll
    for (int mi = 0; mi < 4; ++mi)
        #pragma unroll
        for (int ni = 0; ni < 2; ++ni)
            #pragma unroll
            for (int e = 0; e < 8; ++e) acc[mi][ni][e] = 0.0f;

    const int aoff = half ? 8 : 0;    // A frag K offsets {aoff, aoff+16}
    const int boff = half ? 8 : 0;    // B frag K offsets {boff, boff+16}

#if HAVE_TDM
    if (wave == 0) {                  // pre-issue tile 0 into buffer 0
        unsigned long long ga = (unsigned long long)A  + ((size_t)m0 * K) * 2;
        unsigned long long gb = (unsigned long long)Bw + ((size_t)n0 * K) * 2;
        tdm_load_2d(lds_off(sA[0]), ga, 32, 128, (unsigned long long)K);
        tdm_load_2d(lds_off(sB[0]), gb, 32, 128, (unsigned long long)K);
    }
#endif

    for (int kt = 0; kt < NT; ++kt) {
        const int cur = kt & 1;
        __syncthreads();              // everyone done reading buf[cur^1]
#if HAVE_TDM
        if (wave == 0) {
            if (kt + 1 < NT) {        // prefetch next tile into the other buffer
                unsigned long long ga = (unsigned long long)A  + ((size_t)m0 * K + (size_t)(kt + 1) * 32) * 2;
                unsigned long long gb = (unsigned long long)Bw + ((size_t)n0 * K + (size_t)(kt + 1) * 32) * 2;
                tdm_load_2d(lds_off(sA[cur ^ 1]), ga, 32, 128, (unsigned long long)K);
                tdm_load_2d(lds_off(sB[cur ^ 1]), gb, 32, 128, (unsigned long long)K);
                __builtin_amdgcn_s_wait_tensorcnt(2);   // in-order: current tile done
            } else {
                __builtin_amdgcn_s_wait_tensorcnt(0);
            }
        }
#else
        {   // fallback: manual staging
            const int k0 = kt * 32;
            #pragma unroll
            for (int p = 0; p < 2; ++p) {
                int c = tid + p * 256;
                int r = c >> 2, kc = (c & 3) * 8;
                *(f4raw*)&sA[cur][r * 32 + kc] = *(const f4raw*)&A[(size_t)(m0 + r) * K + k0 + kc];
                *(f4raw*)&sB[cur][r * 32 + kc] = *(const f4raw*)&Bw[(size_t)(n0 + r) * K + k0 + kc];
            }
        }
#endif
        __syncthreads();

        FragU a[4], b[2];
        #pragma unroll
        for (int mi = 0; mi < 4; ++mi) {
            const unsigned short* row = &sA[cur][(wm * 64 + mi * 16 + l16) * 32];
            a[mi].f4[0] = *(const f4raw*)&row[aoff];
            a[mi].f4[1] = *(const f4raw*)&row[aoff + 16];
        }
        #pragma unroll
        for (int ni = 0; ni < 2; ++ni) {
            const unsigned short* row = &sB[cur][(wn * 32 + ni * 16 + l16) * 32];
            b[ni].f4[0] = *(const f4raw*)&row[boff];
            b[ni].f4[1] = *(const f4raw*)&row[boff + 16];
        }
        #pragma unroll
        for (int mi = 0; mi < 4; ++mi)
            #pragma unroll
            for (int ni = 0; ni < 2; ++ni)
                acc[mi][ni] = __builtin_amdgcn_wmma_f32_16x16x32_bf16(
                    false, a[mi].v, false, b[ni].v, (short)0, acc[mi][ni], false, false);
    }

    #pragma unroll
    for (int mi = 0; mi < 4; ++mi)
        #pragma unroll
        for (int ni = 0; ni < 2; ++ni)
            #pragma unroll
            for (int i = 0; i < 8; ++i) {
                int row = m0 + wm * 64 + mi * 16 + i + half * 8;   // C layout: vgpr i -> M
                int col = n0 + wn * 32 + ni * 16 + l16;            //           lane  -> N
                float v = acc[mi][ni][i];
                if (OUT_F32) ((float*)C)[(size_t)row * N + col] = v;
                else ((unsigned short*)C)[(size_t)row * N + col] = f32_to_bf16(v);
            }
}

// ---------------------------------------------------------------------------
// Flash attention (causal). 1 block = (b,h, 64-row Q tile), 128 thr = 4 waves,
// each wave owns 16 Q rows. Q/K tiles staged by TDM; V staged transposed by
// VALU (TDM cannot transpose); online softmax; all matmuls bf16 WMMA f32 acc.
// ---------------------------------------------------------------------------
__global__ __launch_bounds__(128)
void flash_attn(const unsigned short* __restrict__ Q,
                const unsigned short* __restrict__ Kb,
                const unsigned short* __restrict__ V,
                unsigned short* __restrict__ O) {
    __shared__ alignas(16) unsigned short sQ[64 * 64];    // [q][hd]
    __shared__ alignas(16) unsigned short sK[64 * 64];    // [key][hd] == B layout for Q*K^T
    __shared__ alignas(16) unsigned short sVT[64 * 64];   // [hd][key] == B layout for P*V
    __shared__ alignas(16) unsigned short sP[4 * 16 * 64];// per-wave P (A-layout staging)

    const int QT = SEQ / 64;                  // 32 query tiles
    const int qt = blockIdx.x % QT;
    const int bh = blockIdx.x / QT;
    const int h = bh % N_HEADS, b = bh / N_HEADS;
    const int tid = threadIdx.x, lane = tid & 31, wave = tid >> 5;
    const int half = lane >> 4, l16 = lane & 15;
    const int q0 = qt * 64;
    const size_t baseRow = (size_t)b * SEQ;   // rows in the [8192,1024] bf16 buffers
    const int aoff = half ? 8 : 0, boff = half ? 8 : 0;

    // Stage Q tile (64 x 64 bf16, row stride D_MODEL)
#if HAVE_TDM
    if (wave == 0) {
        unsigned long long gq = (unsigned long long)Q +
            ((baseRow + q0) * D_MODEL + (size_t)h * HEAD_DIM) * 2;
        tdm_load_2d(lds_off(sQ), gq, 64, 64, (unsigned long long)D_MODEL);
        __builtin_amdgcn_s_wait_tensorcnt(0);
    }
#else
    #pragma unroll
    for (int p = 0; p < 4; ++p) {
        int c = tid + p * 128;
        int r = c >> 3, ch = (c & 7) * 8;
        *(f4raw*)&sQ[r * 64 + ch] =
            *(const f4raw*)&Q[(baseRow + q0 + r) * D_MODEL + h * HEAD_DIM + ch];
    }
#endif
    __syncthreads();

    FragU qf[2];                              // Q 16x64 -> two K=32 A-fragments
    #pragma unroll
    for (int c = 0; c < 2; ++c) {
        const unsigned short* row = &sQ[(wave * 16 + l16) * 64 + c * 32];
        qf[c].f4[0] = *(const f4raw*)&row[aoff];
        qf[c].f4[1] = *(const f4raw*)&row[aoff + 16];
    }

    v8f o[4]; float m[8], l[8];
    #pragma unroll
    for (int j = 0; j < 4; ++j)
        #pragma unroll
        for (int e = 0; e < 8; ++e) o[j][e] = 0.0f;
    #pragma unroll
    for (int i = 0; i < 8; ++i) { m[i] = -__builtin_inff(); l[i] = 0.0f; }

    const float scale = 0.125f;               // 1/sqrt(64)

    for (int kt = 0; kt <= qt; ++kt) {
        const int k0 = kt * 64;
        __syncthreads();                       // previous tile fully consumed
#if HAVE_TDM
        if (wave == 0) {                       // K tile via TDM
            unsigned long long gk = (unsigned long long)Kb +
                ((baseRow + k0) * D_MODEL + (size_t)h * HEAD_DIM) * 2;
            tdm_load_2d(lds_off(sK), gk, 64, 64, (unsigned long long)D_MODEL);
        }
#else
        #pragma unroll
        for (int p = 0; p < 4; ++p) {
            int c = tid + p * 128;
            int r = c >> 3, ch = (c & 7) * 8;
            *(f4raw*)&sK[r * 64 + ch] =
                *(const f4raw*)&Kb[(baseRow + k0 + r) * D_MODEL + h * HEAD_DIM + ch];
        }
#endif
        #pragma unroll
        for (int p = 0; p < 4; ++p) {          // V tile transposed -> [hd][key]
            int c = tid + p * 128;
            int sr = c >> 3, sc0 = (c & 7) * 8;
            f4raw d = *(const f4raw*)&V[(baseRow + k0 + sr) * D_MODEL + h * HEAD_DIM + sc0];
            const unsigned short* e = (const unsigned short*)&d;
            #pragma unroll
            for (int i = 0; i < 8; ++i) sVT[(sc0 + i) * 64 + sr] = e[i];
        }
        if (kt + 1 <= qt) {                    // hint next V tile toward the caches
            __builtin_prefetch(&V[(baseRow + k0 + 64) * D_MODEL + h * HEAD_DIM + (tid & 63) * 8], 0, 1);
        }
#if HAVE_TDM
        if (wave == 0) __builtin_amdgcn_s_wait_tensorcnt(0);
#endif
        __syncthreads();

        // S = Q * K^T  (16x64 per wave; 4 N-tiles x 2 K-chunks = 8 WMMAs)
        v8f s[4];
        #pragma unroll
        for (int j = 0; j < 4; ++j)
            #pragma unroll
            for (int e = 0; e < 8; ++e) s[j][e] = 0.0f;
        #pragma unroll
        for (int c = 0; c < 2; ++c)
            #pragma unroll
            for (int j = 0; j < 4; ++j) {
                FragU bf;
                const unsigned short* row = &sK[(j * 16 + l16) * 64 + c * 32];
                bf.f4[0] = *(const f4raw*)&row[boff];
                bf.f4[1] = *(const f4raw*)&row[boff + 16];
                s[j] = __builtin_amdgcn_wmma_f32_16x16x32_bf16(
                    false, qf[c].v, false, bf.v, (short)0, s[j], false, false);
            }

        // scale + causal mask (diagonal tile only)
        #pragma unroll
        for (int j = 0; j < 4; ++j)
            #pragma unroll
            for (int i = 0; i < 8; ++i) {
                float v = s[j][i] * scale;
                if (kt == qt) {
                    int qrow = q0 + wave * 16 + i + half * 8;
                    int kcol = k0 + j * 16 + l16;
                    if (kcol > qrow) v = -__builtin_inff();
                }
                s[j][i] = v;
            }

        // row max across the 4 N-tiles + the 16 lanes of this half
        float rmax[8];
        #pragma unroll
        for (int i = 0; i < 8; ++i) {
            float r = s[0][i];
            #pragma unroll
            for (int j = 1; j < 4; ++j) r = fmaxf(r, s[j][i]);
            #pragma unroll
            for (int d = 1; d < 16; d <<= 1) r = fmaxf(r, __shfl_xor(r, d, 32));
            rmax[i] = r;
        }

        float alpha[8];
        #pragma unroll
        for (int i = 0; i < 8; ++i) {
            float mn = fmaxf(m[i], rmax[i]);
            alpha[i] = __expf(m[i] - mn);
            m[i] = mn;
        }
        #pragma unroll
        for (int j = 0; j < 4; ++j)
            #pragma unroll
            for (int i = 0; i < 8; ++i) o[j][i] *= alpha[i];

        float rsum[8];
        #pragma unroll
        for (int i = 0; i < 8; ++i) rsum[i] = 0.0f;
        #pragma unroll
        for (int j = 0; j < 4; ++j)
            #pragma unroll
            for (int i = 0; i < 8; ++i) {
                float p = __expf(s[j][i] - m[i]);
                s[j][i] = p;
                rsum[i] += p;
            }
        #pragma unroll
        for (int i = 0; i < 8; ++i) {
            float r = rsum[i];
            #pragma unroll
            for (int d = 1; d < 16; d <<= 1) r += __shfl_xor(r, d, 32);
            l[i] = l[i] * alpha[i] + r;
        }

        // P (C layout) -> LDS -> A-layout fragments for P*V
        #pragma unroll
        for (int j = 0; j < 4; ++j)
            #pragma unroll
            for (int i = 0; i < 8; ++i)
                sP[wave * 1024 + (i + half * 8) * 64 + j * 16 + l16] = f32_to_bf16(s[j][i]);
        // wave-private region; compiler inserts the ds waits for the RAW below

        #pragma unroll
        for (int c = 0; c < 2; ++c) {
            FragU pf;
            const unsigned short* prow = &sP[wave * 1024 + l16 * 64 + c * 32];
            pf.f4[0] = *(const f4raw*)&prow[aoff];
            pf.f4[1] = *(const f4raw*)&prow[aoff + 16];
            #pragma unroll
            for (int j = 0; j < 4; ++j) {
                FragU vf;
                const unsigned short* vrow = &sVT[(j * 16 + l16) * 64 + c * 32];
                vf.f4[0] = *(const f4raw*)&vrow[boff];
                vf.f4[1] = *(const f4raw*)&vrow[boff + 16];
                o[j] = __builtin_amdgcn_wmma_f32_16x16x32_bf16(
                    false, pf.v, false, vf.v, (short)0, o[j], false, false);
            }
        }
    }

    // normalize + store bf16 into [B*T, D] row-major (ready for the Wo GEMM)
    #pragma unroll
    for (int j = 0; j < 4; ++j)
        #pragma unroll
        for (int i = 0; i < 8; ++i) {
            float v = (l[i] > 0.0f) ? o[j][i] / l[i] : 0.0f;
            int row = q0 + wave * 16 + i + half * 8;
            int col = h * HEAD_DIM + j * 16 + l16;
            O[(baseRow + row) * D_MODEL + col] = f32_to_bf16(v);
        }
}

// ---------------------------------------------------------------------------
// Host-side orchestration (graph-capture safe: launches only)
// ---------------------------------------------------------------------------
extern "C" void kernel_launch(void* const* d_in, const int* in_sizes, int n_in,
                              void* d_out, int out_size, void* d_ws, size_t ws_size,
                              hipStream_t stream) {
    (void)in_sizes; (void)n_in; (void)out_size; (void)ws_size;
    const float* x  = (const float*)d_in[0];
    const float* Wq = (const float*)d_in[1];
    const float* Wk = (const float*)d_in[2];
    const float* Wv = (const float*)d_in[3];
    const float* Wo = (const float*)d_in[4];
    float* out = (float*)d_out;

    char* ws = (char*)d_ws;
    const size_t XB = (size_t)MTOT * D_MODEL * 2;      // 16 MB bf16 activations
    const size_t WB = (size_t)D_MODEL * D_MODEL * 2;   // 2 MB bf16 weights
    unsigned short* xb  = (unsigned short*)(ws);
    unsigned short* wqb = (unsigned short*)(ws + XB);
    unsigned short* wkb = (unsigned short*)(ws + XB + WB);
    unsigned short* wvb = (unsigned short*)(ws + XB + 2 * WB);
    unsigned short* wob = (unsigned short*)(ws + XB + 3 * WB);
    unsigned short* qb  = (unsigned short*)(ws + XB + 4 * WB);
    unsigned short* kb  = (unsigned short*)(ws + 2 * XB + 4 * WB);
    unsigned short* vb  = (unsigned short*)(ws + 3 * XB + 4 * WB);
    unsigned short* ob  = (unsigned short*)(ws + 4 * XB + 4 * WB);

    // Stage 0: casts
    {
        int nx4 = MTOT * D_MODEL / 4, nw4 = D_MODEL * D_MODEL / 4;
        cast_f32_bf16<<<(nx4 + 255) / 256, 256, 0, stream>>>(x,  xb,  nx4);
        cast_f32_bf16<<<(nw4 + 255) / 256, 256, 0, stream>>>(Wq, wqb, nw4);
        cast_f32_bf16<<<(nw4 + 255) / 256, 256, 0, stream>>>(Wk, wkb, nw4);
        cast_f32_bf16<<<(nw4 + 255) / 256, 256, 0, stream>>>(Wv, wvb, nw4);
        cast_f32_bf16<<<(nw4 + 255) / 256, 256, 0, stream>>>(Wo, wob, nw4);
    }

    // Stage 1: Q/K/V projections (bf16 out)
    dim3 ggrid(D_MODEL / 128, MTOT / 128);   // (8, 64)
    gemm_bf16_wmma<false><<<ggrid, 256, 0, stream>>>(xb, wqb, qb, MTOT, D_MODEL, D_MODEL);
    gemm_bf16_wmma<false><<<ggrid, 256, 0, stream>>>(xb, wkb, kb, MTOT, D_MODEL, D_MODEL);
    gemm_bf16_wmma<false><<<ggrid, 256, 0, stream>>>(xb, wvb, vb, MTOT, D_MODEL, D_MODEL);

    // Stage 2: causal flash attention
    flash_attn<<<BATCH * N_HEADS * (SEQ / 64), 128, 0, stream>>>(qb, kb, vb, ob);

    // Stage 3: output projection (f32 out)
    gemm_bf16_wmma<true><<<ggrid, 256, 0, stream>>>(ob, wob, out, MTOT, D_MODEL, D_MODEL);
}
